// Dual_net_33337536151895
// MI455X (gfx1250) — matrix operand order repaired and checked by reading, not compiled, not run on hardware
//
#include <hip/hip_runtime.h>
#include <hip/hip_bf16.h>
#include <math.h>

// ---------------------------------------------------------------------------
// Problem constants (from reference)
// ---------------------------------------------------------------------------
#define N_SUB   54
#define HID     128
#define F_DIM   57          // 3 (case features) + 54 (sub features)
#define K_PAD   64          // F_DIM padded to 2x WMMA K=32 steps
#define BATCH   131072
#define N_COLS  162         // 54 * 3

typedef __attribute__((ext_vector_type(16))) _Float16 v16h;
typedef __attribute__((ext_vector_type(8)))  _Float16 v8h;
typedef __attribute__((ext_vector_type(8)))  float    v8f;

union V16 { v16h v; v8h h[2]; };

// ---------------------------------------------------------------------------
// Kernel 1: convert ow1 (54 x 57 x 128 f32) -> f16 fragments, K padded to 64,
// pre-swizzled into WMMA B-matrix per-lane layout:
//   frag[n][t][kb][lane][e], t = N-tile (h/16), kb = K-block (k/32)
//   lane<16 : N = lane,    K = kb*32 + e          (e = 0..15)
//   lane>=16: N = lane-16, K = kb*32 + 16 + e
// Each lane's 16 halves are contiguous (32 bytes) -> 2x ds_load_b128 later.
// ---------------------------------------------------------------------------
__global__ void k_prep_w(const float* __restrict__ ow1,
                         _Float16* __restrict__ wfrag) {
    const int idx = blockIdx.x * 256 + threadIdx.x;     // 54 * 8192 total
    if (idx >= N_SUB * 8192) return;
    const int n    = idx / 8192;
    const int r    = idx & 8191;
    const int e    = r & 15;
    const int lane = (r >> 4) & 31;
    const int kb   = (r >> 9) & 1;
    const int t    = r >> 10;
    const int col  = lane & 15;
    const int kk   = ((lane >> 4) << 4) | e;            // K offset within 32
    const int k    = kb * 32 + kk;
    const int h    = t * 16 + col;
    float v = (k < F_DIM) ? ow1[((size_t)n * F_DIM + k) * HID + h] : 0.0f;
    wfrag[idx] = (_Float16)v;
}

// ---------------------------------------------------------------------------
// Kernel 2: per-row feature MLPs -> fh[b][64] f16 (row-major, zero padded).
// cf = tanh(case@cw1+cb1)@cw2+cb2 ; nf = tanh(xna@nw1+nb1)@nw2+nb2
// Weights staged in LDS (~60 KB / workgroup).
// ---------------------------------------------------------------------------
__global__ __launch_bounds__(256) void k_features(
    const float* __restrict__ x,
    const float* __restrict__ cw1, const float* __restrict__ cb1,
    const float* __restrict__ cw2, const float* __restrict__ cb2,
    const float* __restrict__ nw1, const float* __restrict__ nb1,
    const float* __restrict__ nw2, const float* __restrict__ nb2,
    _Float16* __restrict__ fh) {

    __shared__ float s_nw1[N_SUB * HID];   // 27648 B
    __shared__ float s_nw2[HID * N_SUB];   // 27648 B
    __shared__ float s_cw1[4 * HID];
    __shared__ float s_cw2[HID * 3];
    __shared__ float s_cb1[HID];
    __shared__ float s_nb1[HID];

    const int tid = threadIdx.x;
    for (int i = tid; i < N_SUB * HID; i += 256) { s_nw1[i] = nw1[i]; s_nw2[i] = nw2[i]; }
    for (int i = tid; i < 4 * HID;     i += 256) s_cw1[i] = cw1[i];
    for (int i = tid; i < HID * 3;     i += 256) s_cw2[i] = cw2[i];
    if (tid < HID) { s_cb1[tid] = cb1[tid]; s_nb1[tid] = nb1[tid]; }
    __syncthreads();

    const long b = (long)blockIdx.x * 256 + tid;
    const float* xr = x + b * 58;

    float cs[4];
    #pragma unroll
    for (int i = 0; i < 4; i++) cs[i] = xr[i];
    float xa[N_SUB];
    #pragma unroll
    for (int i = 0; i < N_SUB; i++) xa[i] = xr[4 + i];

    float cf0 = cb2[0], cf1 = cb2[1], cf2 = cb2[2];
    float nf[N_SUB];
    #pragma unroll
    for (int k = 0; k < N_SUB; k++) nf[k] = nb2[k];

    for (int j = 0; j < HID; j++) {
        // case branch
        float hc = cs[0] * s_cw1[j] + cs[1] * s_cw1[HID + j] +
                   cs[2] * s_cw1[2 * HID + j] + cs[3] * s_cw1[3 * HID + j] + s_cb1[j];
        hc = tanhf(hc);
        cf0 += hc * s_cw2[j * 3 + 0];
        cf1 += hc * s_cw2[j * 3 + 1];
        cf2 += hc * s_cw2[j * 3 + 2];
        // sub-component branch
        float acc = s_nb1[j];
        #pragma unroll
        for (int i = 0; i < N_SUB; i++) acc += xa[i] * s_nw1[i * HID + j];
        float hn = tanhf(acc);
        #pragma unroll
        for (int k = 0; k < N_SUB; k++) nf[k] += hn * s_nw2[j * N_SUB + k];
    }

    _Float16* fr = fh + b * K_PAD;
    fr[0] = (_Float16)cf0; fr[1] = (_Float16)cf1; fr[2] = (_Float16)cf2;
    #pragma unroll
    for (int k = 0; k < N_SUB; k++) fr[3 + k] = (_Float16)nf[k];
    #pragma unroll
    for (int k = F_DIM; k < K_PAD; k++) fr[k] = (_Float16)0.0f;
}

// ---------------------------------------------------------------------------
// Kernel 3: WMMA core. grid = (BATCH/64, 54), block = 128 (4 waves).
// Each wave owns a 16-row batch tile; block shares one n's weight fragments.
//   h[16x128] = tanh(f[16x64] @ ow1h[64x128] + ob1)       -> 16x v_wmma f16
//   v[16x3]   = sigmoid(h @ ow2[128x3] + ob2), normalize  -> VALU + shfl_xor
// ---------------------------------------------------------------------------
__global__ __launch_bounds__(128) void k_gemm(
    const _Float16* __restrict__ fh,
    const _Float16* __restrict__ wfrag,
    const float* __restrict__ ob1,
    const float* __restrict__ ow2,
    const float* __restrict__ ob2,
    float* __restrict__ out) {

    __shared__ __align__(16) _Float16 lds_w[8192];      // 16 KB: B fragments for this n
    __shared__ float  lds_ow2[HID * 3];                 // 1.5 KB
    __shared__ float  lds_h[4][16][HID];                // 32 KB: per-wave h tiles

    const int n    = blockIdx.y;
    const int tid  = threadIdx.x;
    const int wave = tid >> 5;
    const int lane = tid & 31;
    const int m    = lane & 15;   // row within 16-row tile / N within N-tile
    const int hi   = lane >> 4;   // lane half

    const long row0 = (long)blockIdx.x * 64 + wave * 16;
    const _Float16* arow = fh + (row0 + m) * K_PAD;
    __builtin_prefetch(arow, 0, 1);                     // global_prefetch_b8

    // stage this n's B fragments + ow2 into LDS
    const uint4* wsrc = (const uint4*)(wfrag + (size_t)n * 8192);
    uint4* wdst = (uint4*)lds_w;
    #pragma unroll
    for (int i = 0; i < 8; i++) wdst[tid + 128 * i] = wsrc[tid + 128 * i];
    for (int i = tid; i < HID * 3; i += 128) lds_ow2[i] = ow2[n * HID * 3 + i];
    __syncthreads();

    // A fragments: contiguous K order matching B. lane<16: K {0..15}; lane>=16: {16..31}
    V16 a[2];
    #pragma unroll
    for (int kb = 0; kb < 2; kb++) {
        a[kb].h[0] = *(const v8h*)(arow + kb * 32 + hi * 16);
        a[kb].h[1] = *(const v8h*)(arow + kb * 32 + hi * 16 + 8);
    }

    v8f c[8];
    #pragma unroll
    for (int t = 0; t < 8; t++) {
        v8f acc = {};
        #pragma unroll
        for (int kb = 0; kb < 2; kb++) {
            V16 bb;
            const _Float16* bp = lds_w + ((t * 2 + kb) * 32 + lane) * 16;
            bb.h[0] = *(const v8h*)(bp);
            bb.h[1] = *(const v8h*)(bp + 8);
            acc = __builtin_amdgcn_wmma_f32_16x16x32_f16(
                false, a[kb].v, false, bb.v, (short)0, acc, false, false);
        }
        c[t] = acc;
    }

    // bias + tanh, scatter h tile to LDS (C layout: VGPR r -> M = r + 8*hi, N = 16t+m)
    #pragma unroll
    for (int t = 0; t < 8; t++) {
        const int col = t * 16 + m;
        const float bias = ob1[n * HID + col];
        #pragma unroll
        for (int r = 0; r < 8; r++)
            lds_h[wave][r + 8 * hi][col] = tanhf(c[t][r] + bias);
    }
    __asm__ volatile("s_wait_dscnt 0" ::: "memory");    // wave-private LDS RAW fence

    // second GEMM (128 -> 3): half-wave split over H, shfl_xor(16) combine
    float a0 = 0.f, a1 = 0.f, a2 = 0.f;
    const float* hrow = &lds_h[wave][m][hi * 64];
    const float* w2   = &lds_ow2[hi * 64 * 3];
    #pragma unroll 8
    for (int h = 0; h < 64; h++) {
        const float hv = hrow[h];
        a0 += hv * w2[h * 3 + 0];
        a1 += hv * w2[h * 3 + 1];
        a2 += hv * w2[h * 3 + 2];
    }
    a0 += __shfl_xor(a0, 16, 32);
    a1 += __shfl_xor(a1, 16, 32);
    a2 += __shfl_xor(a2, 16, 32);

    if (hi == 0) {
        float v0 = 1.f / (1.f + expf(-(a0 + ob2[n * 3 + 0])));
        float v1 = 1.f / (1.f + expf(-(a1 + ob2[n * 3 + 1])));
        float v2 = 1.f / (1.f + expf(-(a2 + ob2[n * 3 + 2])));
        const float inv = 1.f / (v0 + v1 + v2);
        float* o = out + (row0 + m) * N_COLS + n * 3;
        o[0] = v0 * inv; o[1] = v1 * inv; o[2] = v2 * inv;
    }
}

// ---------------------------------------------------------------------------
// Kernel 4: epilogue. One thread per row, in-place on d_out.
// The 100/sum(p) factor of recon() cancels inside every segment renorm, so:
//   y[j]     = p0'[j]/sum(p0'[0:27]) * c7      (j<27)   etc.
// All 162 v-reads are buffered in registers before any write (in-place safe).
// ---------------------------------------------------------------------------
__global__ __launch_bounds__(256) void k_finalize(const float* __restrict__ x,
                                                  float* __restrict__ out) {
    const long b = (long)blockIdx.x * 256 + threadIdx.x;
    const float* xr = x + b * 58;
    float* orow = out + b * N_COLS;
    const float c7 = xr[1];
    const float c6 = xr[3];

    float p0[N_SUB], p1[N_SUB], p2[N_SUB];
    #pragma unroll
    for (int j = 0; j < N_SUB; j++) {
        const float xa = xr[4 + j];
        float v0 = orow[3 * j + 0];
        float v1 = orow[3 * j + 1];
        float v2 = orow[3 * j + 2];
        if (j == 0)  v0 = 0.f;   // SP_ZERO_COL 0
        if (j == 16) v2 = 0.f;   // SP_ZERO_COL 50
        if (j == 26) v2 = 1.f;   // SP_ONE_COL  80
        p0[j] = xa * v0; p1[j] = xa * v1; p2[j] = xa * v2;
    }
    p0[5] = 0.f;   // ZERO_COL 5
    p1[6] = 0.f;   // ZERO_COL 60  (local 60-54)
    p2[2] = 0.f;   // ZERO_COL 110 (local 110-108)

    float sA0 = 0, sB0 = 0, sA1 = 0, sB1 = 0, s2 = 0;
    #pragma unroll
    for (int j = 0; j < 27; j++)  { sA0 += p0[j]; sA1 += p1[j]; }
    #pragma unroll
    for (int j = 27; j < 54; j++) { sB0 += p0[j]; sB1 += p1[j]; }
    #pragma unroll
    for (int j = 0; j < 54; j++)  s2 += p2[j];

    const float rA0 = c7 / sA0,          rB0 = (100.f - c7) / sB0;
    const float rA1 = c6 / sA1,          rB1 = (100.f - c6) / sB1;
    const float r2  = 100.f / s2;

    #pragma unroll
    for (int j = 0; j < N_SUB; j++) orow[j]       = p0[j] * (j < 27 ? rA0 : rB0);
    #pragma unroll
    for (int j = 0; j < N_SUB; j++) orow[54 + j]  = p1[j] * (j < 27 ? rA1 : rB1);
    #pragma unroll
    for (int j = 0; j < N_SUB; j++) orow[108 + j] = p2[j] * r2;
}

// ---------------------------------------------------------------------------
// Host launch
// ---------------------------------------------------------------------------
extern "C" void kernel_launch(void* const* d_in, const int* in_sizes, int n_in,
                              void* d_out, int out_size, void* d_ws, size_t ws_size,
                              hipStream_t stream) {
    const float* x   = (const float*)d_in[0];
    const float* cw1 = (const float*)d_in[1];
    const float* cb1 = (const float*)d_in[2];
    const float* cw2 = (const float*)d_in[3];
    const float* cb2 = (const float*)d_in[4];
    const float* nw1 = (const float*)d_in[5];
    const float* nb1 = (const float*)d_in[6];
    const float* nw2 = (const float*)d_in[7];
    const float* nb2 = (const float*)d_in[8];
    const float* ow1 = (const float*)d_in[9];
    const float* ob1 = (const float*)d_in[10];
    const float* ow2 = (const float*)d_in[11];
    const float* ob2 = (const float*)d_in[12];
    float* out = (float*)d_out;

    _Float16* fh    = (_Float16*)d_ws;                               // B*64 f16 = 16 MB
    _Float16* wfrag = (_Float16*)((char*)d_ws +
                        (size_t)BATCH * K_PAD * sizeof(_Float16));   // 54*8192 f16

    k_prep_w<<<(N_SUB * 8192) / 256, 256, 0, stream>>>(ow1, wfrag);
    k_features<<<BATCH / 256, 256, 0, stream>>>(x, cw1, cb1, cw2, cb2,
                                                nw1, nb1, nw2, nb2, fh);
    dim3 g(BATCH / 64, N_SUB);
    k_gemm<<<g, 128, 0, stream>>>(fh, wfrag, ob1, ow2, ob2, out);
    k_finalize<<<BATCH / 256, 256, 0, stream>>>(x, out);
}
